// MultiheadAttention_51599737094510
// MI455X (gfx1250) — compile-verified, matrix operand order checked by reading, not run on hardware
//
#include <hip/hip_runtime.h>
#include <hip/hip_bf16.h>

// Problem constants (match reference)
#define BB 4
#define TT 2048
#define CC 2048
#define HH 16
#define GGRP 4
#define HKVN 4
#define DDIM 128
#define KVD 512

typedef __bf16 bf16;
typedef __attribute__((ext_vector_type(16))) __bf16 v16bf;
typedef __attribute__((ext_vector_type(8))) float v8f;
typedef __attribute__((ext_vector_type(4))) int v4i;

// CDNA5 async global->LDS copy path (ASYNCcnt-tracked), if toolchain exposes it
#if defined(__has_builtin)
#if __has_builtin(__builtin_amdgcn_global_load_async_to_lds_b128) && \
    __has_builtin(__builtin_amdgcn_s_wait_asynccnt)
#define HAVE_ASYNC_COPY 1
#endif
#endif
#ifndef HAVE_ASYNC_COPY
#define HAVE_ASYNC_COPY 0
#endif

#if HAVE_ASYNC_COPY
typedef __attribute__((address_space(1))) v4i GASv4i;  // global (prints as __device__)
typedef __attribute__((address_space(3))) v4i LASv4i;  // LDS    (prints as __shared__)
#endif

__device__ __forceinline__ int kbase16(int j, int half) {
    return (j < 4 ? 2 * j : 16 + 2 * (j - 4)) + 8 * half;
}

// A-fragment (16x32, MxK) or B-fragment where memory is [n][k] row-major.
__device__ __forceinline__ v16bf frag_rowmajor(const bf16* base, int ld, int lane) {
    const int lr = lane & 15, half = lane >> 4;
    const bf16* p = base + (size_t)lr * ld;
    v16bf f;
#pragma unroll
    for (int j = 0; j < 8; ++j) {
        int k = kbase16(j, half);
        f[2 * j]     = p[k];
        f[2 * j + 1] = p[k + 1];
    }
    return f;
}

// B-fragment (32x16, KxN) where memory is [k][n] row-major.
__device__ __forceinline__ v16bf frag_kmajor(const bf16* base, int ld, int lane) {
    const int lr = lane & 15, half = lane >> 4;
    v16bf f;
#pragma unroll
    for (int j = 0; j < 8; ++j) {
        int k = kbase16(j, half);
        f[2 * j]     = base[(size_t)k * ld + lr];
        f[2 * j + 1] = base[(size_t)(k + 1) * ld + lr];
    }
    return f;
}

__device__ __forceinline__ v8f wmma_bf16(v16bf a, v16bf b, v8f c) {
    return __builtin_amdgcn_wmma_f32_16x16x32_bf16(
        /*neg_a=*/false, a, /*neg_b=*/false, b,
        /*c_mod=*/(short)0, c, /*reuse_a=*/false, /*reuse_b=*/false);
}

// ---------------------------------------------------------------------------
// Tiled GEMM: Y[M,N] = X[M,K] @ W[N,K]^T + bias[N]
// Block tile 128(M) x 128(N), K-chunks of 32. 256 threads = 8 waves (4Mx2N),
// each wave computes 32x64 = 8 WMMA accumulators per chunk.
// Converts inputs to bf16 while filling LDS. M%128==0, N%128==0, K%32==0.
// ---------------------------------------------------------------------------
template <typename TX, bool OUT_F32>
__global__ __launch_bounds__(256) void gemm_xwT_kernel(
    const TX* __restrict__ X, const float* __restrict__ W,
    const float* __restrict__ bias, void* __restrict__ Yv,
    int M, int N, int K) {
    __shared__ __align__(16) bf16 sX[128][40];
    __shared__ __align__(16) bf16 sW[128][40];

    const int tid  = threadIdx.x;
    const int lane = tid & 31;
    const int wid  = tid >> 5;
    const int wm   = wid >> 1;   // 0..3 -> 32 rows each
    const int wn   = wid & 1;    // 0..1 -> 64 cols each
    const int lr   = lane & 15, half = lane >> 4;
    const int m0   = blockIdx.y * 128;
    const int n0   = blockIdx.x * 128;

    v8f acc[2][4] = {};

    for (int k0 = 0; k0 < K; k0 += 32) {
        __syncthreads();
#pragma unroll
        for (int i = 0; i < 16; ++i) {          // 128x32 tiles of X and W
            int flat = i * 256 + tid;
            int r = flat >> 5, c = flat & 31;
            sX[r][c] = (bf16)(float)X[(size_t)(m0 + r) * K + (k0 + c)];
            sW[r][c] = (bf16)W[(size_t)(n0 + r) * K + (k0 + c)];
        }
        __syncthreads();

        v16bf a0 = frag_rowmajor(&sX[wm * 32][0], 40, lane);
        v16bf a1 = frag_rowmajor(&sX[wm * 32 + 16][0], 40, lane);
#pragma unroll
        for (int fn = 0; fn < 4; ++fn) {
            v16bf bb = frag_rowmajor(&sW[wn * 64 + fn * 16][0], 40, lane);
            acc[0][fn] = wmma_bf16(a0, bb, acc[0][fn]);
            acc[1][fn] = wmma_bf16(a1, bb, acc[1][fn]);
        }
    }

    // Epilogue: C/D layout is (M = j + 8*half, N = lane%16)
#pragma unroll
    for (int fm = 0; fm < 2; ++fm) {
#pragma unroll
        for (int fn = 0; fn < 4; ++fn) {
            int colb = n0 + wn * 64 + fn * 16 + lr;
            int rowb = m0 + wm * 32 + fm * 16 + 8 * half;
            float bv = bias[colb];
#pragma unroll
            for (int j = 0; j < 8; ++j) {
                float v = acc[fm][fn][j] + bv;
                if constexpr (OUT_F32)
                    ((float*)Yv)[(size_t)(rowb + j) * N + colb] = v;
                else
                    ((bf16*)Yv)[(size_t)(rowb + j) * N + colb] = (bf16)v;
            }
        }
    }
}

// ---------------------------------------------------------------------------
// Flash attention (causal, GQA): one block = 64 query rows of one (b, head).
// 128 threads = 4 waves; wave w owns rows [tile*64 + w*16, +16).
// KV tiles (32x128 bf16) double-buffered in LDS, filled by async global->LDS
// copies (ASYNCcnt) when available. Online softmax in f32.
// ---------------------------------------------------------------------------
__global__ __launch_bounds__(128) void attn_flash_kernel(
    const bf16* __restrict__ xq, const bf16* __restrict__ xk,
    const bf16* __restrict__ xv, bf16* __restrict__ xo) {
    const int bh   = blockIdx.x;          // b * HH + qh
    const int b    = bh / HH;
    const int qh   = bh % HH;
    const int hkv  = qh % HKVN;           // reshape (G, HKV): kv head = qh % HKV
    const int tile = blockIdx.y;
    const int tid  = threadIdx.x;
    const int lane = tid & 31;
    const int wid  = tid >> 5;
    const int lr   = lane & 15, half = lane >> 4;
    const int qrow0 = tile * 64 + wid * 16;
    const int seg  = tid & 15;            // 16 segments of 8 bf16 per 128-wide row
    const int r4   = tid >> 4;            // 0..7

    __shared__ __align__(16) bf16 sK[2][32][136];
    __shared__ __align__(16) bf16 sV[2][32][136];
    __shared__ __align__(16) bf16 sP[4][16][40];

    // Issue one 32x128 K tile + V tile into LDS buffer `bufi`.
    // 8 copy instructions per wave per tile (4 rows x {K,V}).
    auto issue_tile = [&](int st, int bufi) {
        const int s0 = st * 32;
#pragma unroll
        for (int i = 0; i < 4; ++i) {
            int r = r4 + i * 8;
            size_t rowoff = ((size_t)(b * TT + s0 + r)) * KVD + hkv * DDIM + seg * 8;
#if HAVE_ASYNC_COPY
            __builtin_amdgcn_global_load_async_to_lds_b128(
                (GASv4i*)(xk + rowoff), (LASv4i*)&sK[bufi][r][seg * 8], 0, 0);
            __builtin_amdgcn_global_load_async_to_lds_b128(
                (GASv4i*)(xv + rowoff), (LASv4i*)&sV[bufi][r][seg * 8], 0, 0);
#else
            *reinterpret_cast<uint4*>(&sK[bufi][r][seg * 8]) =
                *reinterpret_cast<const uint4*>(xk + rowoff);
            *reinterpret_cast<uint4*>(&sV[bufi][r][seg * 8]) =
                *reinterpret_cast<const uint4*>(xv + rowoff);
#endif
        }
    };

    // Preload Q fragments: A-layout 16x32 chunks over D=128 (4 chunks)
    v16bf qf[4];
    {
        const bf16* qb = xq + ((size_t)(b * TT + qrow0 + lr)) * CC + qh * DDIM;
#pragma unroll
        for (int dc = 0; dc < 4; ++dc) {
#pragma unroll
            for (int j = 0; j < 8; ++j) {
                int k = dc * 32 + kbase16(j, half);
                qf[dc][2 * j]     = qb[k];
                qf[dc][2 * j + 1] = qb[k + 1];
            }
        }
    }

    v8f o[8] = {};            // 16x128 output accumulator (8 N-frags)
    float mrow[8], lrow[8];   // per-row running max / sumexp (rows = 8*half + j)
#pragma unroll
    for (int j = 0; j < 8; ++j) { mrow[j] = -3.0e38f; lrow[j] = 0.0f; }

    const float SCALE = 0.08838834764831843f;  // 1/sqrt(128)
    const int nT = 2 * tile + 2;               // causal: s tiles of 32 up to row block end

    issue_tile(0, 0);

    for (int st = 0; st < nT; ++st) {
        const int s0 = st * 32;
        const int cur = st & 1;

        if (st + 1 < nT) {
            issue_tile(st + 1, (st + 1) & 1);   // prefetch next tile into other buffer
#if HAVE_ASYNC_COPY
            __builtin_amdgcn_s_wait_asynccnt(8);  // 8 outstanding = next tile only
#endif
        } else {
#if HAVE_ASYNC_COPY
            __builtin_amdgcn_s_wait_asynccnt(0);
#endif
        }
        __syncthreads();  // (A) tile `st` visible to all waves

        // Wave-uniform causal skip (keeps EXEC all-1s inside WMMA region)
        const bool active = (s0 <= qrow0 + 15);

        if (active) {
            // Scores: S(16x32) = Q(16x128) @ K^T(128x32), two 16x16 C-frags
            v8f sf[2];
#pragma unroll
            for (int sc = 0; sc < 2; ++sc) {
                v8f t = {};
#pragma unroll
                for (int dc = 0; dc < 4; ++dc) {
                    v16bf bk = frag_rowmajor(&sK[cur][sc * 16][dc * 32], 136, lane);
                    t = wmma_bf16(qf[dc], bk, t);
                }
                sf[sc] = t;
            }

            // Online softmax. Fragment row M = qrow0 + 8*half + j, col = s0 + sc*16 + lr
            float p0[8], p1[8], cand[8];
            const int rowb = qrow0 + 8 * half;
            const int col0 = s0 + lr, col1 = s0 + 16 + lr;
#pragma unroll
            for (int j = 0; j < 8; ++j) {
                int row = rowb + j;
                p0[j] = (col0 <= row) ? sf[0][j] * SCALE : -3.0e38f;
                p1[j] = (col1 <= row) ? sf[1][j] * SCALE : -3.0e38f;
                cand[j] = fmaxf(p0[j], p1[j]);
            }
#pragma unroll
            for (int off = 1; off < 16; off <<= 1) {
#pragma unroll
                for (int j = 0; j < 8; ++j)
                    cand[j] = fmaxf(cand[j], __shfl_xor(cand[j], off, 16));
            }
            float resc[8], rs[8];
#pragma unroll
            for (int j = 0; j < 8; ++j) {
                float mn = fmaxf(mrow[j], cand[j]);
                resc[j] = __expf(mrow[j] - mn);
                mrow[j] = mn;
                p0[j] = __expf(p0[j] - mn);
                p1[j] = __expf(p1[j] - mn);
                rs[j] = p0[j] + p1[j];
            }
#pragma unroll
            for (int off = 1; off < 16; off <<= 1) {
#pragma unroll
                for (int j = 0; j < 8; ++j)
                    rs[j] += __shfl_xor(rs[j], off, 16);
            }
#pragma unroll
            for (int j = 0; j < 8; ++j) lrow[j] = lrow[j] * resc[j] + rs[j];
#pragma unroll
            for (int f = 0; f < 8; ++f) {
#pragma unroll
                for (int j = 0; j < 8; ++j) o[f][j] *= resc[j];
            }
            // Dump P (C-layout) to per-wave LDS scratch for A-layout reload
#pragma unroll
            for (int j = 0; j < 8; ++j) {
                sP[wid][8 * half + j][lr]      = (bf16)p0[j];
                sP[wid][8 * half + j][16 + lr] = (bf16)p1[j];
            }
        }
        __syncthreads();  // (B) P visible; V tile still valid

        if (active) {
            // O += P(16x32) @ V(32x128)
            v16bf pa = frag_rowmajor(&sP[wid][0][0], 40, lane);
#pragma unroll
            for (int dv = 0; dv < 8; ++dv) {
                v16bf bv = frag_kmajor(&sV[cur][0][dv * 16], 136, lane);
                o[dv] = wmma_bf16(pa, bv, o[dv]);
            }
        }
        __syncthreads();  // (C) compute done before buffer `cur` is re-issued
    }

    // Normalize and write xo (B,T,C), c = qh*D + d
#pragma unroll
    for (int j = 0; j < 8; ++j) lrow[j] = 1.0f / lrow[j];
#pragma unroll
    for (int f = 0; f < 8; ++f) {
#pragma unroll
        for (int j = 0; j < 8; ++j) {
            int row = qrow0 + 8 * half + j;
            xo[((size_t)(b * TT + row)) * CC + qh * DDIM + f * 16 + lr] =
                (bf16)(o[f][j] * lrow[j]);
        }
    }
}

// ---------------------------------------------------------------------------

extern "C" void kernel_launch(void* const* d_in, const int* in_sizes, int n_in,
                              void* d_out, int out_size, void* d_ws, size_t ws_size,
                              hipStream_t stream) {
    const float* query = (const float*)d_in[0];
    const float* key   = (const float*)d_in[1];
    const float* value = (const float*)d_in[2];
    const float* ipw   = (const float*)d_in[3];  // (C+2KV, C)
    const float* ipb   = (const float*)d_in[4];  // (C+2KV,)
    const float* opw   = (const float*)d_in[5];  // (C, C)
    const float* opb   = (const float*)d_in[6];  // (C,)
    float* out = (float*)d_out;

    // Workspace: xq (bf16 B*T*C), xk/xv (bf16 B*T*KV), xo (bf16 B*T*C) ~84MB
    bf16* xq = (bf16*)d_ws;
    bf16* xk = xq + (size_t)BB * TT * CC;
    bf16* xv = xk + (size_t)BB * TT * KVD;
    bf16* xo = xv + (size_t)BB * TT * KVD;

    const int M = BB * TT;  // 8192
    dim3 blk(256);

    // Q / K / V projections (f32 in -> bf16 out, converted in LDS fill)
    gemm_xwT_kernel<float, false><<<dim3(CC / 128, M / 128), blk, 0, stream>>>(
        query, ipw, ipb, xq, M, CC, CC);
    gemm_xwT_kernel<float, false><<<dim3(KVD / 128, M / 128), blk, 0, stream>>>(
        key, ipw + (size_t)CC * CC, ipb + CC, xk, M, KVD, CC);
    gemm_xwT_kernel<float, false><<<dim3(KVD / 128, M / 128), blk, 0, stream>>>(
        value, ipw + (size_t)(CC + KVD) * CC, ipb + CC + KVD, xv, M, KVD, CC);

    // Flash attention: grid (B*H heads, T/64 row tiles), 4 waves per block
    attn_flash_kernel<<<dim3(BB * HH, TT / 64), dim3(128), 0, stream>>>(xq, xk, xv, xo);

    // Output projection (bf16 in -> f32 out to d_out)
    gemm_xwT_kernel<bf16, true><<<dim3(CC / 128, M / 128), blk, 0, stream>>>(
        xo, opw, opb, out, M, CC, CC);
}
